// UnifiedPhysicsModel_90254442758611
// MI455X (gfx1250) — matrix operand order repaired and checked by reading, not compiled, not run on hardware
//
#include <hip/hip_runtime.h>
#include <hip/hip_bf16.h>
#include <stdint.h>

typedef __attribute__((ext_vector_type(16))) __bf16 v16bf;
typedef __attribute__((ext_vector_type(8)))  float  v8f;

#define PI_F        3.14159265358979323846f
#define TWO_PI_F    6.28318530717958647692f
#define INV_2PI_F   0.15915494309189533577f

// ---------------------------------------------------------------- WMMA helpers
__device__ __forceinline__ v8f wmma_bf16(v16bf a, v16bf b, v8f c) {
  return __builtin_amdgcn_wmma_f32_16x16x32_bf16(false, a, false, b, (short)0, c,
                                                 false, false);
}

struct bf16x16 { uint4 lo, hi; };

// A fragment (16x32 bf16 tile, row-major, this lane's row): two contiguous
// 16-byte spans at K offsets [16h, 16h+7] and [16h+8, 16h+15].
__device__ __forceinline__ v16bf load_frag_a(const __bf16* p, int hf) {
  bf16x16 r;
  r.lo = *reinterpret_cast<const uint4*>(p + 16 * hf);
  r.hi = *reinterpret_cast<const uint4*>(p + 16 * hf + 8);
  return __builtin_bit_cast(v16bf, r);
}

// B fragment pre-packed to [frag][lane][16]: one contiguous 32-byte load.
__device__ __forceinline__ v16bf load_frag_b(const __bf16* p) {
  const uint4* q = reinterpret_cast<const uint4*>(p);
  bf16x16 r; r.lo = q[0]; r.hi = q[1];
  return __builtin_bit_cast(v16bf, r);
}

__device__ __forceinline__ float wave_sum(float v) {
#pragma unroll
  for (int off = 16; off > 0; off >>= 1) v += __shfl_xor(v, off, 32);
  return v;
}

// ------------------------------------------------- weight prep: pack B frags
// dst layout: [K/32][N/16][lane 0..31][e 0..15]
// frag element e of lane l  <->  B[kc*32 + 16*(l>>4) + e][ct*16 + (l&15)]
__global__ void pack_b(const float* __restrict__ src, __bf16* __restrict__ dst,
                       int srcK, int K, int N) {
  unsigned i = blockIdx.x * blockDim.x + threadIdx.x;
  unsigned total = (unsigned)K * N;
  if (i >= total) return;
  int e = i & 15;
  int lane = (i >> 4) & 31;
  unsigned fi = i >> 9;
  int colTiles = N >> 4;
  int kc = fi / colTiles, ct = fi % colTiles;
  int k = kc * 32 + (lane >> 4) * 16 + e;
  int n = ct * 16 + (lane & 15);
  dst[i] = (k < srcK) ? (__bf16)src[(size_t)k * N + n] : (__bf16)0.f;
}

// ---------------------------------------------------------------- generic GEMM
// C[M,N](f32) = A[M,K](bf16 row-major) * Bpacked[K,N] + bias
// One wave computes a 16x64 strip (4 accumulators); (N/16) % 4 == 0.
__global__ void gemm_bf16(const __bf16* __restrict__ A,
                          const __bf16* __restrict__ Bp,
                          const float* __restrict__ bias, float* __restrict__ C,
                          int M, int N, int K) {
  int wave = threadIdx.x >> 5, lane = threadIdx.x & 31;
  int hf = lane >> 4, mn = lane & 15;
  int colTiles = N >> 4, colGroups = colTiles >> 2;
  int g = blockIdx.x * (blockDim.x >> 5) + wave;
  if (g >= (M >> 4) * colGroups) return;
  int rt = g / colGroups, cg = g % colGroups;
  const __bf16* Ab = A + (size_t)(rt * 16 + mn) * K;
  const __bf16* Bf = Bp + ((size_t)(cg * 4) * 32 + lane) * 16;
  size_t bstep = (size_t)colTiles * 32 * 16;  // one K-chunk of fragments
  v8f acc[4] = {{}, {}, {}, {}};
  for (int k0 = 0; k0 < K; k0 += 32) {
    v16bf a = load_frag_a(Ab + k0, hf);
#pragma unroll
    for (int t = 0; t < 4; ++t) {
      v16bf b = load_frag_b(Bf + (size_t)t * (32 * 16));
      acc[t] = wmma_bf16(a, b, acc[t]);
    }
    Bf += bstep;
  }
#pragma unroll
  for (int t = 0; t < 4; ++t) {
    int ct = cg * 4 + t;
    float bv = bias ? bias[ct * 16 + mn] : 0.f;
    float* Cp = C + (size_t)(rt * 16 + hf * 8) * N + ct * 16 + mn;
#pragma unroll
    for (int r = 0; r < 8; ++r) Cp[(size_t)r * N] = acc[t][r] + bv;
  }
}

// ------------------------------------------------- implicit-GEMM conv (WMMA)
// up==0: stride-2 SAME conv   (iy = 2*oy + ky)
// up==1: stride-2 conv-transpose SAME (iy = (oy + ky - 1)/2, parity gated)
// One wave computes 16 output pixels x 64 output channels; (Cout/16) % 4 == 0.
__global__ void conv_wmma(const __bf16* __restrict__ X,
                          const __bf16* __restrict__ Bp,
                          const float* __restrict__ bias, float* __restrict__ Y,
                          int Hin, int Win, int Cin, int Hout, int Wout,
                          int Cout, int up) {
  int wave = threadIdx.x >> 5, lane = threadIdx.x & 31;
  int hf = lane >> 4, mn = lane & 15;
  int rows = 32 * Hout * Wout;
  int colTiles = Cout >> 4, colGroups = colTiles >> 2;
  int g = blockIdx.x * (blockDim.x >> 5) + wave;
  if (g >= (rows >> 4) * colGroups) return;
  int rt = g / colGroups, cg = g % colGroups;
  int grow = rt * 16 + mn;
  int img = grow / (Hout * Wout);
  int rem = grow % (Hout * Wout);
  int oy = rem / Wout, ox = rem % Wout;
  int K = 9 * Cin;
  const __bf16* Bf = Bp + ((size_t)(cg * 4) * 32 + lane) * 16;
  size_t bstep = (size_t)colTiles * 32 * 16;
  v8f acc[4] = {{}, {}, {}, {}};
  for (int k0 = 0; k0 < K; k0 += 32) {
    int tap = k0 / Cin, cbase = k0 % Cin;
    int ky = tap / 3, kx = tap % 3;
    int iy, ix;
    bool valid;
    if (!up) {
      iy = oy * 2 + ky; ix = ox * 2 + kx;
      valid = (iy < Hin) && (ix < Win);
    } else {
      int ty = oy + ky - 1, tx = ox + kx - 1;
      valid = (ty >= 0) && (tx >= 0) && ((ty & 1) == 0) && ((tx & 1) == 0);
      iy = ty >> 1; ix = tx >> 1;
      valid = valid && (iy < Hin) && (ix < Win);
    }
    v16bf a = {};
    if (valid)
      a = load_frag_a(X + ((((size_t)img * Hin + iy) * Win + ix) * Cin + cbase),
                      hf);
#pragma unroll
    for (int t = 0; t < 4; ++t) {
      v16bf b = load_frag_b(Bf + (size_t)t * (32 * 16));
      acc[t] = wmma_bf16(a, b, acc[t]);
    }
    Bf += bstep;
  }
#pragma unroll
  for (int t = 0; t < 4; ++t) {
    int ct = cg * 4 + t;
    float bv = bias[ct * 16 + mn];
    float* Yp = Y + (size_t)(rt * 16 + hf * 8) * Cout + ct * 16 + mn;
#pragma unroll
    for (int r = 0; r < 8; ++r) Yp[(size_t)r * Cout] = acc[t][r] + bv;
  }
}

// ----------------------------------------- embedding gather + projection GEMM
__global__ void embed_project(const int* cc, const int* nb, const int* il,
                              const int* in_, const int* ip, const int* iw,
                              const float* Ec, const float* Enb,
                              const float* El, const float* En,
                              const float* Ep, const float* Ew,
                              const __bf16* __restrict__ Bp /*packed 224x256*/,
                              const float* bias, __bf16* X) {
  __shared__ __attribute__((aligned(16))) __bf16 As[16 * 224];
  int rt = blockIdx.x, tid = threadIdx.x;
  for (int e = tid; e < 16 * 224; e += 256) {
    int r = e / 224, k = e % 224;
    int g = rt * 16 + r;
    float v = 0.f;
    if (k < 64)       v = Ec[(cc[g] % 300) * 64 + k];
    else if (k < 80)  v = Enb[nb[g] * 16 + (k - 64)];
    else if (k < 112) v = El[il[g] * 32 + (k - 80)];
    else if (k < 144) v = En[in_[g] * 32 + (k - 112)];
    else if (k < 176) v = Ep[ip[g] * 32 + (k - 144)];
    else if (k < 208) v = Ew[iw[g] * 32 + (k - 176)];
    As[e] = (__bf16)v;
  }
  __syncthreads();
  int wave = tid >> 5, lane = tid & 31, hf = lane >> 4, mn = lane & 15;
  int ct0 = wave * 2;
  const __bf16* Bf = Bp + ((size_t)ct0 * 32 + lane) * 16;
  v8f acc0 = {}, acc1 = {};
#pragma unroll
  for (int k0 = 0; k0 < 224; k0 += 32) {
    v16bf a = load_frag_a(&As[mn * 224 + k0], hf);
    v16bf b0 = load_frag_b(Bf);
    v16bf b1 = load_frag_b(Bf + 32 * 16);
    acc0 = wmma_bf16(a, b0, acc0);
    acc1 = wmma_bf16(a, b1, acc1);
    Bf += (size_t)16 * 32 * 16;  // 16 colTiles per K-chunk
  }
#pragma unroll
  for (int t = 0; t < 2; ++t) {
    int ct = ct0 + t;
    float bv = bias[ct * 16 + mn];
    __bf16* Xp = X + ((size_t)rt * 16 + hf * 8) * 256 + ct * 16 + mn;
    v8f& acc = t ? acc1 : acc0;
#pragma unroll
    for (int r = 0; r < 8; ++r) Xp[(size_t)r * 256] = (__bf16)(acc[r] + bv);
  }
}

// ---------------------------------------------------------------- LN + gelu
__global__ void ln_gelu(const float* __restrict__ Y, const float* s,
                        const float* b, __bf16* Ob, float* Of, int rows, int C) {
  int row = blockIdx.x * (blockDim.x >> 5) + (threadIdx.x >> 5);
  if (row >= rows) return;
  int lane = threadIdx.x & 31;
  const float* y = Y + (size_t)row * C;
  float sum = 0.f, sq = 0.f;
  for (int c = lane; c < C; c += 32) { float v = y[c]; sum += v; sq += v * v; }
  sum = wave_sum(sum); sq = wave_sum(sq);
  float m = sum / C;
  float rstd = rsqrtf(sq / C - m * m + 1e-6f);
  for (int c = lane; c < C; c += 32) {
    float v = (y[c] - m) * rstd * s[c] + b[c];
    float g = 0.5f * v * (1.f + tanhf(0.7978845608f * (v + 0.044715f * v * v * v)));
    if (Ob) Ob[(size_t)row * C + c] = (__bf16)g;
    if (Of) Of[(size_t)row * C + c] = g;
  }
}

// ---------------------------------------------------------------- world state
__global__ void init_state(const float* per, float* pr, int* pq, float* o,
                           float* tr, int* tq) {
  int t = blockIdx.x * (blockDim.x >> 5) + (threadIdx.x >> 5);
  if (t >= 1024) return;
  int lane = threadIdx.x & 31;
  float sq = 0.f;
  for (int d = lane; d < 256; d += 32) { float v = per[(size_t)t * 256 + d]; sq += v * v; }
  sq = wave_sum(sq);
  float inv = 1.f / (sqrtf(sq) + 1e-9f);
  for (int d = lane; d < 256; d += 32) {
    size_t i = (size_t)t * 256 + d;
    float v = per[i];
    float tpi = v + PI_F, q = floorf(tpi * INV_2PI_F);
    pr[i] = tpi - TWO_PI_F * q - PI_F;
    pq[i] = 0; o[i] = v * inv; tr[i] = 0.f; tq[i] = 0;
  }
}

__global__ void step_intra(const float* pr, float* intra, float* inter) {
  int b = blockIdx.x, d = threadIdx.x;
  float accb = 0.f;
  for (int p = 0; p < 4; ++p) {
    float s = 0.f;
    for (int n = 0; n < 32; ++n)
      s += pr[((size_t)((b * 4 + p) * 32) + n) * 256 + d];
    s *= (1.f / 32.f);
    intra[(b * 4 + p) * 256 + d] = s;
    accb += s;
  }
  inter[b * 256 + d] = accb * 0.25f;
}

__global__ void step_fuse(const float* pr, const float* o, const float* tr,
                          const float* intra, const float* inter, float* f,
                          __bf16* up_, __bf16* uo_, __bf16* ut_) {
  int bp = blockIdx.x, b = bp >> 2;
  __shared__ float ti[256], frs[32], wsm[32], red[8];
  __shared__ float tin;
  int tid = threadIdx.x, wave = tid >> 5, lane = tid & 31;
  ti[tid] = intra[bp * 256 + tid] + inter[b * 256 + tid];
  __syncthreads();
  float v = ti[tid];
  float sq = wave_sum(v * v);
  if (lane == 0) red[wave] = sq;
  __syncthreads();
  if (tid == 0) { float s = 0; for (int i = 0; i < 8; ++i) s += red[i]; tin = sqrtf(s); }
  __syncthreads();
  float n2 = tin;
  for (int n = wave; n < 32; n += 8) {
    const float* op = o + ((size_t)bp * 32 + n) * 256;
    float dot = 0.f, no2 = 0.f;
    for (int d = lane; d < 256; d += 32) { float ov = op[d]; dot += ov * ti[d]; no2 += ov * ov; }
    dot = wave_sum(dot); no2 = wave_sum(no2);
    if (lane == 0) {
      float n1 = sqrtf(no2);
      frs[n] = (n1 > 1e-8f && n2 > 1e-8f) ? dot / (n1 * n2 + 1e-8f) : 0.f;
    }
  }
  __syncthreads();
  if (wave == 0) {
    float x = frs[lane], mx = x;
#pragma unroll
    for (int off = 16; off > 0; off >>= 1) mx = fmaxf(mx, __shfl_xor(mx, off, 32));
    float e = expf(x - mx);
    float s = wave_sum(e);
    wsm[lane] = e / s;
  }
  __syncthreads();
  for (int idx = tid; idx < 32 * 256; idx += 256) {
    int n = idx >> 8, d = idx & 255;
    float fv = wsm[n] * ti[d];
    size_t gi = ((size_t)bp * 32 + n) * 256 + d;
    f[gi] = fv;
    up_[gi] = (__bf16)(pr[gi] + fv);
    uo_[gi] = (__bf16)(o[gi] + fv);
    ut_[gi] = (__bf16)(tr[gi] + fv);
  }
}

__global__ void gate_softmax(const __bf16* u, const float* gate, float* g) {
  int t = blockIdx.x * (blockDim.x >> 5) + (threadIdx.x >> 5);
  if (t >= 1024) return;
  int lane = threadIdx.x & 31;
  float acc0 = 0, acc1 = 0, acc2 = 0, acc3 = 0;
  for (int d = lane; d < 256; d += 32) {
    float uv = (float)u[(size_t)t * 256 + d];
    acc0 += uv * gate[d * 4 + 0]; acc1 += uv * gate[d * 4 + 1];
    acc2 += uv * gate[d * 4 + 2]; acc3 += uv * gate[d * 4 + 3];
  }
  acc0 = wave_sum(acc0); acc1 = wave_sum(acc1);
  acc2 = wave_sum(acc2); acc3 = wave_sum(acc3);
  if (lane == 0) {
    float mx = fmaxf(fmaxf(acc0, acc1), fmaxf(acc2, acc3));
    float e0 = expf(acc0 - mx), e1 = expf(acc1 - mx);
    float e2 = expf(acc2 - mx), e3 = expf(acc3 - mx);
    float s = e0 + e1 + e2 + e3;
    g[t * 4 + 0] = e0 / s; g[t * 4 + 1] = e1 / s;
    g[t * 4 + 2] = e2 / s; g[t * 4 + 3] = e3 / s;
  }
}

__global__ void grammar_combine(const float* base1, const float* f,
                                const float* g, const float* e, __bf16* out) {
  unsigned idx = blockIdx.x * blockDim.x + threadIdx.x;  // 262144
  int t = idx >> 8, d = idx & 255;
  float u = base1[idx] + f[idx];
  float acc = u;
#pragma unroll
  for (int k = 0; k < 4; ++k)
    acc += g[t * 4 + k] * e[(size_t)t * 1024 + k * 256 + d];
  out[idx] = (__bf16)acc;
}

__global__ void step_update(float* pr, int* pq, float* o, float* tr, int* tq,
                            const float* dp, const float* dov, const float* dt) {
  int t = blockIdx.x, d = threadIdx.x;
  int wave = d >> 5, lane = d & 31;
  __shared__ float red[8];
  __shared__ float nrm;
  size_t i = (size_t)t * 256 + d;
  float u = pr[i] + dp[i];
  float tpi = u + PI_F, q = floorf(tpi * INV_2PI_F);
  pr[i] = tpi - TWO_PI_F * q - PI_F;
  pq[i] += (int)q;
  float od = o[i] + dov[i];
  float sq = wave_sum(od * od);
  if (lane == 0) red[wave] = sq;
  __syncthreads();
  if (d == 0) { float s = 0; for (int k = 0; k < 8; ++k) s += red[k]; nrm = sqrtf(s) + 1e-9f; }
  __syncthreads();
  o[i] = od / nrm;
  float ut = tr[i] + dt[i];
  float tpi2 = ut + PI_F, q2 = floorf(tpi2 * INV_2PI_F);
  tr[i] = tpi2 - TWO_PI_F * q2 - PI_F;
  tq[i] += (int)q2;
}

__global__ void energy_kernel(const float* tr, const int* tq, float* out) {
  int bp = blockIdx.x, tid = threadIdx.x, wave = tid >> 5, lane = tid & 31;
  __shared__ float red[8];
  float acc = 0.f;
  for (int j = tid; j < 8192; j += 256) {
    size_t i = (size_t)bp * 8192 + j;
    acc += (1.f - cosf(tr[i])) + 0.001f * fabsf((float)tq[i]);
  }
  acc = wave_sum(acc);
  if (lane == 0) red[wave] = acc;
  __syncthreads();
  if (tid == 0) { float s = 0; for (int k = 0; k < 8; ++k) s += red[k]; out[bp] = s; }
}

// -------------------------- dec_pre: (32x8192)@(8192x8192), async-LDS staging
__global__ void dec_pre(const float* __restrict__ x, const float* __restrict__ W,
                        const float* __restrict__ b, float* y, __bf16* ybf) {
  __shared__ __attribute__((aligned(16))) float xs[32 * 64];  // 8 KB chunk
  int col = blockIdx.x * 256 + threadIdx.x;
  float acc[32];
#pragma unroll
  for (int r = 0; r < 32; ++r) acc[r] = 0.f;
  uint32_t xsbase = (uint32_t)(uintptr_t)&xs[0];
  uint64_t gb = (uint64_t)(uintptr_t)x;
  for (int k0 = 0; k0 < 8192; k0 += 64) {
    __syncthreads();  // previous chunk fully consumed
#pragma unroll
    for (int i = 0; i < 2; ++i) {  // 512 x b128 copies, 2 per thread
      int E = threadIdx.x + i * 256;
      int r = E >> 4, k4 = (E & 15) << 2;
      uint32_t lo = xsbase + (uint32_t)((r << 6) + k4) * 4u;
      uint32_t go = (uint32_t)((r * 8192 + k0 + k4) * 4);
      asm volatile("global_load_async_to_lds_b128 %0, %1, %2"
                   :: "v"(lo), "v"(go), "s"(gb) : "memory");
    }
    asm volatile("s_wait_asynccnt 0" ::: "memory");
    __syncthreads();
    __builtin_prefetch(&W[(size_t)(k0 + 64) * 8192 + col], 0, 1);
    for (int kk = 0; kk < 64; ++kk) {
      float w = W[(size_t)(k0 + kk) * 8192 + col];
#pragma unroll
      for (int r = 0; r < 32; ++r) acc[r] += xs[r * 64 + kk] * w;
    }
  }
  float bv = b[col];
#pragma unroll
  for (int r = 0; r < 32; ++r) {
    float v = acc[r] + bv;
    y[(size_t)r * 8192 + col] = v;
    ybf[(size_t)r * 8192 + col] = (__bf16)v;
  }
}

// ---------------------------------------------------------------- bilinear 2x
__global__ void bilinear2x(const float* __restrict__ in, float* __restrict__ out) {
  size_t idx = (size_t)blockIdx.x * blockDim.x + threadIdx.x;  // 67108864
  int c = idx & 255;
  size_t t = idx >> 8;
  int x = (int)(t % 128); t /= 128;
  int y = (int)(t % 64);  t /= 64;
  int img = (int)t;
  float sy = y * 0.5f - 0.25f, sx = x * 0.5f - 0.25f;
  int y0 = (int)floorf(sy), x0 = (int)floorf(sx);
  float fy = sy - y0, fx = sx - x0;
  int y0c = min(max(y0, 0), 31), y1c = min(max(y0 + 1, 0), 31);
  int x0c = min(max(x0, 0), 63), x1c = min(max(x0 + 1, 0), 63);
  const float* base = in + (size_t)img * 32 * 64 * 256 + c;
  float v00 = base[((size_t)y0c * 64 + x0c) * 256];
  float v01 = base[((size_t)y0c * 64 + x1c) * 256];
  float v10 = base[((size_t)y1c * 64 + x0c) * 256];
  float v11 = base[((size_t)y1c * 64 + x1c) * 256];
  out[idx] = (1.f - fy) * ((1.f - fx) * v00 + fx * v01) +
             fy * ((1.f - fx) * v10 + fx * v11);
}

// ================================================================ host driver
enum {
  IN_CC = 0, IN_NB, IN_IL, IN_IN, IN_IP, IN_IW,
  P_EMB_CHAR, P_EMB_NB, P_EMB_L, P_EMB_N, P_EMB_P, P_EMB_W,
  P_PROJ_W, P_PROJ_B,
  P_CK0, P_CB0, P_LS0, P_LB0, P_CK1, P_CB1, P_LS1, P_LB1,
  P_CK2, P_CB2, P_LS2, P_LB2, P_CK3, P_CB3, P_LS3, P_LB3,
  P_POS_GATE, P_POS_GFW, P_POS_GFB, P_ORI_GATE, P_ORI_GFW, P_ORI_GFB,
  P_POS_DW, P_POS_DB, P_ORI_DW, P_ORI_DB, P_TEN_DW, P_TEN_DB,
  P_DEC_W, P_DEC_B,
  P_DK0, P_DB0, P_DLS0, P_DLB0, P_DK1, P_DB1, P_DLS1, P_DLB1,
  P_DK2, P_DB2, P_DLS2, P_DLB2,
  P_OUT_W, P_OUT_B
};

extern "C" void kernel_launch(void* const* d_in, const int* in_sizes, int n_in,
                              void* d_out, int out_size, void* d_ws,
                              size_t ws_size, hipStream_t stream) {
  (void)in_sizes; (void)n_in; (void)out_size; (void)ws_size;
  auto F = [&](int i) { return (const float*)d_in[i]; };
  auto I = [&](int i) { return (const int*)d_in[i]; };

  char* wsp = (char*)d_ws;
  size_t cur = 0;
  auto alloc = [&](size_t bytes) -> void* {
    void* r = wsp + cur;
    cur += (bytes + 255) & ~(size_t)255;
    return r;
  };

  // shared pool: x0 (encoder) aliases decoder buffers
  char* pool = (char*)alloc(134217728);
  __bf16* x0 = (__bf16*)pool;
  size_t pc = 0;
  auto palloc = [&](size_t bytes) -> void* {
    void* r = pool + pc;
    pc += (bytes + 255) & ~(size_t)255;
    return r;
  };
  float*  yfp    = (float*)palloc(1048576);
  __bf16* ybf    = (__bf16*)palloc(524288);
  float*  d0o    = (float*)palloc(4194304);
  __bf16* dl0    = (__bf16*)palloc(2097152);
  float*  d1o    = (float*)palloc(8388608);
  __bf16* dl1    = (__bf16*)palloc(4194304);
  float*  d2o    = (float*)palloc(16777216);
  __bf16* dl2    = (__bf16*)palloc(8388608);
  float*  logits = (float*)palloc(67108864);

  // packed bf16 weight fragments
  __bf16* wb_proj = (__bf16*)alloc(224 * 256 * 2);
  __bf16* wb_c0 = (__bf16*)alloc(2304 * 64 * 2);
  __bf16* wb_c1 = (__bf16*)alloc(576 * 128 * 2);
  __bf16* wb_c2 = (__bf16*)alloc(1152 * 256 * 2);
  __bf16* wb_c3 = (__bf16*)alloc(2304 * 256 * 2);
  __bf16* wb_gfp = (__bf16*)alloc(256 * 1024 * 2);
  __bf16* wb_gfo = (__bf16*)alloc(256 * 1024 * 2);
  __bf16* wb_dp = (__bf16*)alloc(256 * 256 * 2);
  __bf16* wb_do = (__bf16*)alloc(256 * 256 * 2);
  __bf16* wb_dt = (__bf16*)alloc(256 * 256 * 2);
  __bf16* wb_d0 = (__bf16*)alloc(2304 * 256 * 2);
  __bf16* wb_d1 = (__bf16*)alloc(2304 * 128 * 2);
  __bf16* wb_d2 = (__bf16*)alloc(1152 * 64 * 2);
  __bf16* wb_ow = (__bf16*)alloc(64 * 256 * 2);

  // encoder activations
  float*  c0o = (float*)alloc(16777216);
  __bf16* l0  = (__bf16*)alloc(8388608);
  float*  c1o = (float*)alloc(8388608);
  __bf16* l1  = (__bf16*)alloc(4194304);
  float*  c2o = (float*)alloc(4194304);
  __bf16* l2  = (__bf16*)alloc(2097152);
  float*  c3o = (float*)alloc(1048576);
  float*  per = (float*)alloc(1048576);

  // world state + step temporaries
  float* pr = (float*)alloc(1048576);
  int*   pq = (int*)alloc(1048576);
  float* ovec = (float*)alloc(1048576);
  float* tr = (float*)alloc(1048576);
  int*   tq = (int*)alloc(1048576);
  float* intra = (float*)alloc(32768);
  float* inter = (float*)alloc(8192);
  float* fbuf = (float*)alloc(1048576);
  __bf16* u_pos = (__bf16*)alloc(524288);
  __bf16* u_ori = (__bf16*)alloc(524288);
  __bf16* u_ten = (__bf16*)alloc(524288);
  float* g_pos = (float*)alloc(16384);
  float* g_ori = (float*)alloc(16384);
  float* e_pos = (float*)alloc(4194304);
  float* e_ori = (float*)alloc(4194304);
  __bf16* pu = (__bf16*)alloc(524288);
  __bf16* ou = (__bf16*)alloc(524288);
  float* dp = (float*)alloc(1048576);
  float* dov = (float*)alloc(1048576);
  float* dt = (float*)alloc(1048576);

  auto pk = [&](const float* s, __bf16* d, int srcK, int K, int N) {
    unsigned n = (unsigned)K * N;
    pack_b<<<(n + 255) / 256, 256, 0, stream>>>(s, d, srcK, K, N);
  };
  pk(F(P_PROJ_W), wb_proj, 208, 224, 256);
  pk(F(P_CK0), wb_c0, 2304, 2304, 64);
  pk(F(P_CK1), wb_c1, 576, 576, 128);
  pk(F(P_CK2), wb_c2, 1152, 1152, 256);
  pk(F(P_CK3), wb_c3, 2304, 2304, 256);
  pk(F(P_POS_GFW), wb_gfp, 256, 256, 1024);
  pk(F(P_ORI_GFW), wb_gfo, 256, 256, 1024);
  pk(F(P_POS_DW), wb_dp, 256, 256, 256);
  pk(F(P_ORI_DW), wb_do, 256, 256, 256);
  pk(F(P_TEN_DW), wb_dt, 256, 256, 256);
  pk(F(P_DK0), wb_d0, 2304, 2304, 256);
  pk(F(P_DK1), wb_d1, 2304, 2304, 128);
  pk(F(P_DK2), wb_d2, 1152, 1152, 64);
  pk(F(P_OUT_W), wb_ow, 64, 64, 256);

  // ---- encoder
  embed_project<<<16384, 256, 0, stream>>>(
      I(IN_CC), I(IN_NB), I(IN_IL), I(IN_IN), I(IN_IP), I(IN_IW),
      F(P_EMB_CHAR), F(P_EMB_NB), F(P_EMB_L), F(P_EMB_N), F(P_EMB_P),
      F(P_EMB_W), wb_proj, F(P_PROJ_B), x0);

  conv_wmma<<<512, 256, 0, stream>>>(x0, wb_c0, F(P_CB0), c0o, 64, 128, 256, 32, 64, 64, 0);
  ln_gelu<<<8192, 256, 0, stream>>>(c0o, F(P_LS0), F(P_LB0), l0, nullptr, 65536, 64);
  conv_wmma<<<256, 256, 0, stream>>>(l0, wb_c1, F(P_CB1), c1o, 32, 64, 64, 16, 32, 128, 0);
  ln_gelu<<<2048, 256, 0, stream>>>(c1o, F(P_LS1), F(P_LB1), l1, nullptr, 16384, 128);
  conv_wmma<<<128, 256, 0, stream>>>(l1, wb_c2, F(P_CB2), c2o, 16, 32, 128, 8, 16, 256, 0);
  ln_gelu<<<512, 256, 0, stream>>>(c2o, F(P_LS2), F(P_LB2), l2, nullptr, 4096, 256);
  conv_wmma<<<32, 256, 0, stream>>>(l2, wb_c3, F(P_CB3), c3o, 8, 16, 256, 4, 8, 256, 0);
  ln_gelu<<<128, 256, 0, stream>>>(c3o, F(P_LS3), F(P_LB3), nullptr, per, 1024, 256);

  init_state<<<128, 256, 0, stream>>>(per, pr, pq, ovec, tr, tq);

  // ---- 8 dynamics steps
  for (int s = 0; s < 8; ++s) {
    step_intra<<<8, 256, 0, stream>>>(pr, intra, inter);
    step_fuse<<<32, 256, 0, stream>>>(pr, ovec, tr, intra, inter, fbuf, u_pos, u_ori, u_ten);
    gate_softmax<<<128, 256, 0, stream>>>(u_pos, F(P_POS_GATE), g_pos);
    gate_softmax<<<128, 256, 0, stream>>>(u_ori, F(P_ORI_GATE), g_ori);
    gemm_bf16<<<128, 256, 0, stream>>>(u_pos, wb_gfp, F(P_POS_GFB), e_pos, 1024, 1024, 256);
    gemm_bf16<<<128, 256, 0, stream>>>(u_ori, wb_gfo, F(P_ORI_GFB), e_ori, 1024, 1024, 256);
    grammar_combine<<<1024, 256, 0, stream>>>(pr, fbuf, g_pos, e_pos, pu);
    grammar_combine<<<1024, 256, 0, stream>>>(ovec, fbuf, g_ori, e_ori, ou);
    gemm_bf16<<<32, 256, 0, stream>>>(pu, wb_dp, F(P_POS_DB), dp, 1024, 256, 256);
    gemm_bf16<<<32, 256, 0, stream>>>(ou, wb_do, F(P_ORI_DB), dov, 1024, 256, 256);
    gemm_bf16<<<32, 256, 0, stream>>>(u_ten, wb_dt, F(P_TEN_DB), dt, 1024, 256, 256);
    step_update<<<1024, 256, 0, stream>>>(pr, pq, ovec, tr, tq, dp, dov, dt);
  }

  float* out_f = (float*)d_out;
  energy_kernel<<<32, 256, 0, stream>>>(tr, tq, out_f + 67108864);

  // ---- decoder
  dec_pre<<<32, 256, 0, stream>>>(pr, F(P_DEC_W), F(P_DEC_B), yfp, ybf);
  conv_wmma<<<128, 256, 0, stream>>>(ybf, wb_d0, F(P_DB0), d0o, 4, 8, 256, 8, 16, 256, 1);
  ln_gelu<<<512, 256, 0, stream>>>(d0o, F(P_DLS0), F(P_DLB0), dl0, nullptr, 4096, 256);
  conv_wmma<<<256, 256, 0, stream>>>(dl0, wb_d1, F(P_DB1), d1o, 8, 16, 256, 16, 32, 128, 1);
  ln_gelu<<<2048, 256, 0, stream>>>(d1o, F(P_DLS1), F(P_DLB1), dl1, nullptr, 16384, 128);
  conv_wmma<<<512, 256, 0, stream>>>(dl1, wb_d2, F(P_DB2), d2o, 16, 32, 128, 32, 64, 64, 1);
  ln_gelu<<<8192, 256, 0, stream>>>(d2o, F(P_DLS2), F(P_DLB2), dl2, nullptr, 65536, 64);
  gemm_bf16<<<2048, 256, 0, stream>>>(dl2, wb_ow, F(P_OUT_B), logits, 65536, 256, 64);
  bilinear2x<<<262144, 256, 0, stream>>>(logits, out_f);
}
